// MultiHeadAttention_12446815223829
// MI455X (gfx1250) — matrix operand order repaired and checked by reading, not compiled, not run on hardware
//
#include <hip/hip_runtime.h>
#include <cstddef>

// ---------------- CDNA5 WMMA types ----------------
typedef __attribute__((ext_vector_type(16))) __bf16 v16bf;
typedef __attribute__((ext_vector_type(8)))  float  v8f;
typedef __attribute__((ext_vector_type(4)))  int    v4i;

union ABFrag { v16bf v; unsigned int u[8]; };

struct __align__(16) U32x4 { unsigned int x, y, z, w; };
struct __align__(8)  U32x2 { unsigned int x, y; };

#define T_SEQ 4096
#define C_DIM 768
#define NH    12
#define HS    64
#define BATCH 2

// ---- feature gates (compile-safe on both toolchains) ----
#if defined(__HIP_DEVICE_COMPILE__) && defined(__gfx1250__) &&                 \
    __has_builtin(__builtin_amdgcn_global_load_async_to_lds_b128) &&           \
    __has_builtin(__builtin_amdgcn_s_wait_asynccnt)
#define HAVE_ASYNC_LDS 1
#else
#define HAVE_ASYNC_LDS 0
#endif

#if defined(__HIP_DEVICE_COMPILE__) && __has_builtin(__builtin_amdgcn_cvt_pk_bf16_f32)
#define HAVE_PK_BF16 1
#else
#define HAVE_PK_BF16 0
#endif

#if HAVE_ASYNC_LDS
// pointer-to-v4i in global (AS1) / LDS (AS3) address spaces
typedef __attribute__((address_space(1))) v4i* v4i_gptr;
typedef __attribute__((address_space(3))) v4i* v4i_lptr;
#endif

__device__ __forceinline__ unsigned short f2bf(float f) {
  unsigned int u = __float_as_uint(f);
  u += 0x7FFFu + ((u >> 16) & 1u);   // round-to-nearest-even (ignoring NaN edge)
  return (unsigned short)(u >> 16);
}

// pack two fp32 -> packed bf16x2 (one VALU op when v_cvt_pk_bf16_f32 exists)
__device__ __forceinline__ unsigned int f2bf2(float lo, float hi) {
#if HAVE_PK_BF16
  auto r = __builtin_amdgcn_cvt_pk_bf16_f32(lo, hi);
  return __builtin_bit_cast(unsigned int, r);
#else
  return (unsigned int)f2bf(lo) | ((unsigned int)f2bf(hi) << 16);
#endif
}

// 16-byte global->LDS copy: async (ASYNCcnt) when available, else via VGPRs
__device__ __forceinline__ void copy16_g2l(const unsigned short* g, unsigned short* l) {
#if HAVE_ASYNC_LDS
  __builtin_amdgcn_global_load_async_to_lds_b128(
      (v4i_gptr)(void*)g, (v4i_lptr)(void*)l, 0, 0);
#else
  *(U32x4*)l = *(const U32x4*)g;
#endif
}

__device__ __forceinline__ void copy_g2l_wait() {
#if HAVE_ASYNC_LDS
  __builtin_amdgcn_s_wait_asynccnt(0);
#endif
}

// =====================================================================
// Kernel 1: qkv = x @ W_attn^T   (M=8192, N=2304, K=768)
// 128x64 block tile, 4 waves, each wave owns 32x64 (8 wmma / K-step).
// fp32 in -> bf16 LDS tiles -> v_wmma_f32_16x16x32_bf16 -> bf16 scatter
// =====================================================================
__global__ __launch_bounds__(128)
void gemm_qkv_bf16(const float* __restrict__ X, const float* __restrict__ W,
                   unsigned short* __restrict__ qkv) {
  __shared__ unsigned short As[128][32];
  __shared__ unsigned short Bs[64][32];
  const int tid  = threadIdx.x;
  const int lane = tid & 31, wave = tid >> 5;
  const int half = lane >> 4, l16 = lane & 15;
  const int m0 = blockIdx.y << 7;
  const int n0 = blockIdx.x << 6;
  v8f acc[2][4] = {};

  for (int k0 = 0; k0 < C_DIM; k0 += 32) {
    __syncthreads();
#pragma unroll
    for (int i = 0; i < 8; ++i) {              // A tile: 1024 float4
      int f4  = (i << 7) + tid;
      int row = f4 >> 3;
      int kc  = (f4 & 7) << 2;
      float4 a = *(const float4*)&X[(size_t)(m0 + row) * C_DIM + k0 + kc];
      U32x2 av = { f2bf2(a.x, a.y), f2bf2(a.z, a.w) };
      *(U32x2*)&As[row][kc] = av;
    }
#pragma unroll
    for (int i = 0; i < 4; ++i) {              // B tile: 512 float4
      int f4  = (i << 7) + tid;
      int row = f4 >> 3;
      int kc  = (f4 & 7) << 2;
      float4 b = *(const float4*)&W[(size_t)(n0 + row) * C_DIM + k0 + kc];
      U32x2 bv = { f2bf2(b.x, b.y), f2bf2(b.z, b.w) };
      *(U32x2*)&Bs[row][kc] = bv;
    }
    if (k0 + 32 < C_DIM)
      __builtin_prefetch(&X[(size_t)(m0 + tid) * C_DIM + k0 + 32], 0, 0);
    __syncthreads();

    ABFrag af[2];
#pragma unroll
    for (int mt = 0; mt < 2; ++mt)
#pragma unroll
      for (int i = 0; i < 8; ++i) {
        int kk = ((i & 3) << 1) + (half << 3) + ((i >> 2) << 4);
        af[mt].u[i] = *(const unsigned int*)&As[(wave << 5) + (mt << 4) + l16][kk];
      }
#pragma unroll
    for (int nt = 0; nt < 4; ++nt) {
      ABFrag bf;
#pragma unroll
      for (int i = 0; i < 8; ++i)
        bf.u[i] = *(const unsigned int*)&Bs[(nt << 4) + l16][((i & 3) << 1) + (half << 3) + ((i >> 2) << 4)];
#pragma unroll
      for (int mt = 0; mt < 2; ++mt)
        acc[mt][nt] = __builtin_amdgcn_wmma_f32_16x16x32_bf16(
            false, af[mt].v, false, bf.v, (short)0, acc[mt][nt], false, false);
    }
  }

#pragma unroll
  for (int mt = 0; mt < 2; ++mt) {
    const int mbase = m0 + (wave << 5) + (mt << 4);
#pragma unroll
    for (int nt = 0; nt < 4; ++nt) {
      int n     = n0 + (nt << 4) + l16;
      int which = n / C_DIM;
      int rem   = n - which * C_DIM;
      int hh = rem >> 6, d = rem & 63;
#pragma unroll
      for (int r = 0; r < 8; ++r) {
        int m = mbase + r + (half << 3);
        int b = m >> 12, t = m & (T_SEQ - 1);
        size_t idx = ((((size_t)which * BATCH + b) * NH + hh) * T_SEQ + t) * HS + d;
        qkv[idx] = f2bf(acc[mt][nt][r]);
      }
    }
  }
}

// =====================================================================
// Kernel 2: flash attention, one block = (b, h, 64 query rows), 4 waves.
// Q/K tiles staged via async global->LDS when available.
// =====================================================================
__global__ __launch_bounds__(128)
void attn_flash(const unsigned short* __restrict__ qkv,
                unsigned short* __restrict__ att) {
  __shared__ unsigned short Qs[64][64];
  __shared__ unsigned short Ks[64][64];
  __shared__ unsigned short Vts[64][64];     // transposed: [d][key]
  __shared__ unsigned short Ps[4][16][64];   // per-wave P tile

  const int tid  = threadIdx.x;
  const int lane = tid & 31, wave = tid >> 5;
  const int half = lane >> 4, l16 = lane & 15;
  const int bh = blockIdx.x;             // 0..23
  const int b = bh / NH, h = bh % NH;
  const int qi = blockIdx.y;             // 0..63 query tile

  const size_t plane = (size_t)T_SEQ * HS;
  const unsigned short* Qg = qkv + ((size_t)(0 * BATCH + b) * NH + h) * plane;
  const unsigned short* Kg = qkv + ((size_t)(1 * BATCH + b) * NH + h) * plane;
  const unsigned short* Vg = qkv + ((size_t)(2 * BATCH + b) * NH + h) * plane;

  // ---- stage Q tile (async when available) ----
#pragma unroll
  for (int i = 0; i < 4; ++i) {
    int u4  = (i << 7) + tid;           // 8 x 16B per row
    int row = u4 >> 3;
    int dc  = (u4 & 7) << 3;
    copy16_g2l(&Qg[(size_t)((qi << 6) + row) * HS + dc], &Qs[row][dc]);
  }
  copy_g2l_wait();
  __syncthreads();

  ABFrag qf[2];
#pragma unroll
  for (int kc = 0; kc < 2; ++kc)
#pragma unroll
    for (int i = 0; i < 8; ++i) {
      int dd = (kc << 5) + ((i & 3) << 1) + (half << 3) + ((i >> 2) << 4);
      qf[kc].u[i] = *(const unsigned int*)&Qs[(wave << 4) + l16][dd];
    }

  v8f o[4] = {};
  float mrow[8], lrow[8];
#pragma unroll
  for (int r = 0; r < 8; ++r) { mrow[r] = -1e30f; lrow[r] = 0.f; }

  for (int j = 0; j <= qi; ++j) {
    __syncthreads();
    // ---- stage K (async direct) and V (VGPR transpose) tiles ----
#pragma unroll
    for (int i = 0; i < 4; ++i) {
      int u4  = (i << 7) + tid;
      int row = u4 >> 3;
      int dc  = (u4 & 7) << 3;
      copy16_g2l(&Kg[(size_t)((j << 6) + row) * HS + dc], &Ks[row][dc]);
      U32x4 pv = *(const U32x4*)&Vg[(size_t)((j << 6) + row) * HS + dc];
      unsigned int wv[4] = { pv.x, pv.y, pv.z, pv.w };
#pragma unroll
      for (int w = 0; w < 4; ++w) {
        Vts[dc + 2 * w    ][row] = (unsigned short)(wv[w] & 0xFFFFu);
        Vts[dc + 2 * w + 1][row] = (unsigned short)(wv[w] >> 16);
      }
    }
    copy_g2l_wait();
    __syncthreads();

    // ---- S = Q @ K^T  (16x64 per wave) ----
    v8f s[4];
#pragma unroll
    for (int nt = 0; nt < 4; ++nt) {
      v8f sa = {};
#pragma unroll
      for (int kc = 0; kc < 2; ++kc) {
        ABFrag bf;
#pragma unroll
        for (int i = 0; i < 8; ++i)
          bf.u[i] = *(const unsigned int*)&Ks[(nt << 4) + l16][(kc << 5) + ((i & 3) << 1) + (half << 3) + ((i >> 2) << 4)];
        sa = __builtin_amdgcn_wmma_f32_16x16x32_bf16(
            false, qf[kc].v, false, bf.v, (short)0, sa, false, false);
      }
      s[nt] = sa;
    }

    // ---- scale + causal mask ----
    const bool diag = (j == qi);
#pragma unroll
    for (int nt = 0; nt < 4; ++nt)
#pragma unroll
      for (int r = 0; r < 8; ++r) {
        float v = s[nt][r] * 0.125f;   // 1/sqrt(64)
        if (diag) {
          int qr = (wave << 4) + r + (half << 3);
          int kk = (nt << 4) + l16;
          if (kk > qr) v = -1e30f;
        }
        s[nt][r] = v;
      }

    // ---- online softmax ----
    float mnew[8], alpha[8];
#pragma unroll
    for (int r = 0; r < 8; ++r) {
      float v = fmaxf(fmaxf(s[0][r], s[1][r]), fmaxf(s[2][r], s[3][r]));
      v = fmaxf(v, __shfl_xor(v, 1));
      v = fmaxf(v, __shfl_xor(v, 2));
      v = fmaxf(v, __shfl_xor(v, 4));
      v = fmaxf(v, __shfl_xor(v, 8));
      mnew[r]  = fmaxf(mrow[r], v);
      alpha[r] = __expf(mrow[r] - mnew[r]);
      mrow[r]  = mnew[r];
      lrow[r] *= alpha[r];
    }
#pragma unroll
    for (int nt = 0; nt < 4; ++nt)
#pragma unroll
      for (int r = 0; r < 8; ++r) o[nt][r] *= alpha[r];

#pragma unroll
    for (int r = 0; r < 8; ++r) {
      float rs = 0.f;
#pragma unroll
      for (int nt = 0; nt < 4; ++nt) {
        float p = __expf(s[nt][r] - mnew[r]);
        rs += p;
        Ps[wave][r + (half << 3)][(nt << 4) + l16] = f2bf(p);
      }
      rs += __shfl_xor(rs, 1);
      rs += __shfl_xor(rs, 2);
      rs += __shfl_xor(rs, 4);
      rs += __shfl_xor(rs, 8);
      lrow[r] += rs;
    }
    __syncthreads();   // P visible; fences Ks reads vs next overwrite

    // ---- O += P @ V ----
#pragma unroll
    for (int kc = 0; kc < 2; ++kc) {
      ABFrag pf;
#pragma unroll
      for (int i = 0; i < 8; ++i) {
        int kk = (kc << 5) + ((i & 3) << 1) + (half << 3) + ((i >> 2) << 4);
        pf.u[i] = *(const unsigned int*)&Ps[wave][l16][kk];
      }
#pragma unroll
      for (int nt = 0; nt < 4; ++nt) {
        ABFrag vf;
#pragma unroll
        for (int i = 0; i < 8; ++i)
          vf.u[i] = *(const unsigned int*)&Vts[(nt << 4) + l16][(kc << 5) + ((i & 3) << 1) + (half << 3) + ((i >> 2) << 4)];
        o[nt] = __builtin_amdgcn_wmma_f32_16x16x32_bf16(
            false, pf.v, false, vf.v, (short)0, o[nt], false, false);
      }
    }
  }

  // ---- normalize and store bf16 att output (B,T,C) ----
#pragma unroll
  for (int r = 0; r < 8; ++r) {
    float inv = 1.0f / lrow[r];
    int t = (qi << 6) + (wave << 4) + r + (half << 3);
#pragma unroll
    for (int nt = 0; nt < 4; ++nt) {
      int c = (h << 6) + (nt << 4) + l16;
      att[((size_t)b * T_SEQ + t) * C_DIM + c] = f2bf(o[nt][r] * inv);
    }
  }
}

// =====================================================================
// Kernel 3: out = att @ W_proj^T  (M=8192, N=768, K=768), fp32 output
// 128x64 block tile; bf16 A tile staged with async global->LDS.
// =====================================================================
__global__ __launch_bounds__(128)
void gemm_proj_f32(const unsigned short* __restrict__ A, const float* __restrict__ W,
                   float* __restrict__ out) {
  __shared__ unsigned short As[128][32];
  __shared__ unsigned short Bs[64][32];
  const int tid  = threadIdx.x;
  const int lane = tid & 31, wave = tid >> 5;
  const int half = lane >> 4, l16 = lane & 15;
  const int m0 = blockIdx.y << 7;
  const int n0 = blockIdx.x << 6;
  v8f acc[2][4] = {};

  for (int k0 = 0; k0 < C_DIM; k0 += 32) {
    __syncthreads();
#pragma unroll
    for (int i = 0; i < 4; ++i) {              // A tile: 512 x 16B (async path)
      int u4  = (i << 7) + tid;
      int row = u4 >> 2;
      int kc  = (u4 & 3) << 3;
      copy16_g2l(&A[(size_t)(m0 + row) * C_DIM + k0 + kc], &As[row][kc]);
    }
#pragma unroll
    for (int i = 0; i < 4; ++i) {              // B tile: fp32 -> bf16
      int f4  = (i << 7) + tid;
      int row = f4 >> 3;
      int kc  = (f4 & 7) << 2;
      float4 b = *(const float4*)&W[(size_t)(n0 + row) * C_DIM + k0 + kc];
      U32x2 bv = { f2bf2(b.x, b.y), f2bf2(b.z, b.w) };
      *(U32x2*)&Bs[row][kc] = bv;
    }
    copy_g2l_wait();
    __syncthreads();

    ABFrag af[2];
#pragma unroll
    for (int mt = 0; mt < 2; ++mt)
#pragma unroll
      for (int i = 0; i < 8; ++i) {
        int kk = ((i & 3) << 1) + (half << 3) + ((i >> 2) << 4);
        af[mt].u[i] = *(const unsigned int*)&As[(wave << 5) + (mt << 4) + l16][kk];
      }
#pragma unroll
    for (int nt = 0; nt < 4; ++nt) {
      ABFrag bf;
#pragma unroll
      for (int i = 0; i < 8; ++i)
        bf.u[i] = *(const unsigned int*)&Bs[(nt << 4) + l16][((i & 3) << 1) + (half << 3) + ((i >> 2) << 4)];
#pragma unroll
      for (int mt = 0; mt < 2; ++mt)
        acc[mt][nt] = __builtin_amdgcn_wmma_f32_16x16x32_bf16(
            false, af[mt].v, false, bf.v, (short)0, acc[mt][nt], false, false);
    }
  }

#pragma unroll
  for (int mt = 0; mt < 2; ++mt) {
    const int mbase = m0 + (wave << 5) + (mt << 4);
#pragma unroll
    for (int nt = 0; nt < 4; ++nt) {
      int n = n0 + (nt << 4) + l16;
#pragma unroll
      for (int r = 0; r < 8; ++r) {
        int m = mbase + r + (half << 3);
        out[(size_t)m * C_DIM + n] = acc[mt][nt][r];
      }
    }
  }
}

// =====================================================================
extern "C" void kernel_launch(void* const* d_in, const int* in_sizes, int n_in,
                              void* d_out, int out_size, void* d_ws, size_t ws_size,
                              hipStream_t stream) {
  (void)in_sizes; (void)n_in; (void)out_size; (void)ws_size;
  const float* X  = (const float*)d_in[0];   // (B,T,C) fp32
  const float* Wa = (const float*)d_in[1];   // (3C,C) fp32
  const float* Wp = (const float*)d_in[2];   // (C,C)  fp32
  float* out = (float*)d_out;                // (B,T,C) fp32

  unsigned short* qkv = (unsigned short*)d_ws;                       // 3*B*nH*T*hs bf16
  const size_t QKV_ELEMS = (size_t)3 * BATCH * NH * T_SEQ * HS;      // 18,874,368
  unsigned short* att = qkv + QKV_ELEMS;                             // B*T*C bf16

  dim3 blk(128);
  // qkv GEMM: N tiles = 2304/64 = 36, M tiles = 8192/128 = 64
  gemm_qkv_bf16<<<dim3(36, 64), blk, 0, stream>>>(X, Wa, qkv);
  // attention: 24 (b,h) x 64 query tiles
  attn_flash<<<dim3(BATCH * NH, T_SEQ / 64), blk, 0, stream>>>(qkv, att);
  // projection GEMM: N tiles = 768/64 = 12, M tiles = 64
  gemm_proj_f32<<<dim3(12, 64), blk, 0, stream>>>(att, Wp, out);
}
